// Lawformer_21234318312239
// MI455X (gfx1250) — matrix operand order repaired and checked by reading, not compiled, not run on hardware
//
#include <hip/hip_runtime.h>
#include <hip/hip_bf16.h>
#include <math.h>

typedef __bf16 v16bf __attribute__((ext_vector_type(16)));
typedef float  v8f   __attribute__((ext_vector_type(8)));
typedef unsigned short u16;
typedef unsigned int   u32;
typedef unsigned long long u64;

union ABfrag { v16bf v; u16 u[16]; u32 w[8]; };

__device__ __forceinline__ float bf2f(u16 x){ u32 t=((u32)x)<<16; float f; __builtin_memcpy(&f,&t,4); return f; }
__device__ __forceinline__ u16  f2bf(float f){ u32 t; __builtin_memcpy(&t,&f,4); t += 0x7FFFu + ((t>>16)&1u); return (u16)(t>>16); }
__device__ __forceinline__ int  iclamp(int x,int lo,int hi){ return x<lo?lo:(x>hi?hi:x); }

__device__ __forceinline__ v8f wmma_bf(const ABfrag&a, const ABfrag&b, v8f c){
  return __builtin_amdgcn_wmma_f32_16x16x32_bf16(false, a.v, false, b.v, (short)0, c, false, false);
}

// A fragment (16x32 bf16, MxK) from a row-major source at (row0,k0), ld = row stride in elems.
__device__ __forceinline__ void loadA_rm(ABfrag&f, const u16* p, int ld){
  const int lane = threadIdx.x & 31; const int m = lane & 15; const int hi = lane >> 4;
  const u16* r = p + m*ld + hi*16;
  #pragma unroll
  for(int j=0;j<8;j++){ const int kb=2*j; f.w[j] = *(const u32*)(r + kb); }
}
// B fragment (32x16 bf16, KxN) from a [N][K] layout (ld = K stride in elems).
__device__ __forceinline__ void loadB_nk(ABfrag&f, const u16* p, int ld){
  const int lane = threadIdx.x & 31; const int n = lane & 15; const int hi = lane >> 4;
  const u16* r = p + n*ld + hi*16;
  #pragma unroll
  for(int j=0;j<8;j++){ f.w[j] = *(const u32*)(r + 2*j); }
}

// ---------------------------------------------------------------------------
// Tensor Data Mover: build a 2-D D# (count=1, type=image, data_size=2B) and issue
// TENSOR_LOAD_TO_LDS. Per CDNA5 ISA.
#if __has_builtin(__builtin_amdgcn_tensor_load_to_lds)
#define USE_TDM 1
typedef unsigned int v4u __attribute__((ext_vector_type(4)));
typedef int v8i __attribute__((ext_vector_type(8)));
typedef int v4i __attribute__((ext_vector_type(4)));

__device__ __forceinline__ void tdm_load_2d_bf16(
    u32 lds_off, const void* gaddr,
    u32 tile_d0, u32 tile_d1,            // tile X (contig elems), tile Y (rows)
    u32 tensor_d0, u32 tensor_d1,        // extents for OOB (elems)
    u64 row_stride,                      // dim0 stride in elems
    u32 pad_en, u32 pad_interval, u32 pad_amount)
{
  const u64 ga = (u64)gaddr;
  v4u g0;
  g0[0] = 1u;                                            // count=1, user descriptor
  g0[1] = lds_off;                                       // lds_addr (bytes)
  g0[2] = (u32)(ga & 0xFFFFFFFFu);                       // global_addr[31:0]
  g0[3] = (u32)((ga >> 32) & 0x01FFFFFFu) | (2u << 30);  // global_addr[56:32] | type=2
  v8i g1;
  g1[0] = (int)((1u<<16) | (pad_en<<20) | (pad_interval<<22) | (pad_amount<<25)); // data_size=2B
  g1[1] = (int)((tensor_d0 & 0xFFFFu) << 16);            // abar_addr=0 | tensor_dim0 lo
  g1[2] = (int)((tensor_d0 >> 16) | ((tensor_d1 & 0xFFFFu) << 16));
  g1[3] = (int)((tensor_d1 >> 16) | (tile_d0 << 16));
  g1[4] = (int)(tile_d1 & 0xFFFFu);                      // tile_dim1 | tile_dim2=0
  g1[5] = (int)(u32)(row_stride & 0xFFFFFFFFull);        // tensor_dim0_stride[31:0]
  g1[6] = (int)(u32)((row_stride >> 32) & 0xFFFFull);    // stride[47:32] | dim1_stride lo=0
  g1[7] = 0;
  v4i z4 = {0,0,0,0};
#if __clang_major__ >= 23
  v8i z8 = {0,0,0,0,0,0,0,0};
  __builtin_amdgcn_tensor_load_to_lds(g0, g1, z4, z4, z8, 0);
#else
  __builtin_amdgcn_tensor_load_to_lds(g0, g1, z4, z4, 0);
#endif
}
#endif

// ---------------------------------------------------------------------------
// fp32 [K][N] -> bf16 [N][K] transpose-convert (32x32 tiles through LDS).
// Weights are stored transposed so GEMM B tiles have the same row-major
// geometry as A tiles (contiguous K) -> TDM-friendly + ds_load_b128 frags.
__global__ __launch_bounds__(256) void f2bf_tr_k(
    const float* __restrict__ x, u16* __restrict__ y, int K, int N)
{
  __shared__ u16 t[32][33];
  const int tn = N >> 5;
  const int ti = blockIdx.x % tn;      // n tile
  const int tj = blockIdx.x / tn;      // k tile
  const int c  = threadIdx.x & 31, r8 = threadIdx.x >> 5;
  #pragma unroll
  for(int i=0;i<4;i++){
    const int kk = tj*32 + r8 + i*8;   // k row (coalesced read over n)
    const int nn = ti*32 + c;
    t[c][r8 + i*8] = f2bf(x[(size_t)kk*N + nn]);   // t[n_local][k_local]
  }
  __syncthreads();
  #pragma unroll
  for(int i=0;i<4;i++){
    const int nn = ti*32 + r8 + i*8;
    const int kk = tj*32 + c;
    y[(size_t)nn*K + kk] = t[r8 + i*8][c];         // coalesced write over k
  }
}

// ---------------------------------------------------------------------------
// bf16 WMMA GEMM: C[M,N] = A[M,K]*Bt[N,K]^T (+bias). 128x128 block tile, 8 waves
// (4x2), 32-deep K steps. Both tiles are 128 rows x 32 contiguous K elems and are
// staged to LDS by the Tensor Data Mover (double buffered, TENSORcnt-tracked).
// Descriptor LDS-padding gives row stride 40 elems (80 B): 16B-aligned rows
// (b128-mergeable fragment loads) and bank stride 20 (conflict-free half-wave).
// epi: 1 = bias -> f32 Cf; 2 = (bias+acc)*scale -> bf16 Cb; 3 = gelu(bias+acc) -> bf16 Cb.
__global__ __launch_bounds__(256) void gemm_bf16_k(
    const u16* __restrict__ A, const u16* __restrict__ Bt, const float* __restrict__ bias,
    float* __restrict__ Cf, u16* __restrict__ Cb, int M, int N, int K, int epi, float scale)
{
  __shared__ u16 As[2][128][40];   // [m][k]
  __shared__ u16 Bs[2][128][40];   // [n][k]
  const int tid=threadIdx.x, lane=tid&31, wv=tid>>5;
  const int wm=wv>>1, wn=wv&1;
  const int m0=blockIdx.y*128, n0=blockIdx.x*128;
  const v8f zero8 = {0.f,0.f,0.f,0.f,0.f,0.f,0.f,0.f};
  v8f acc[2][4];
  #pragma unroll
  for(int i=0;i<2;i++)
    #pragma unroll
    for(int j=0;j<4;j++) acc[i][j]=zero8;

  const int nk=K>>5;

#ifdef USE_TDM
  auto issue=[&](int kt,int buf){
    // A tile: 32 contiguous k-elems x 128 m-rows, row stride K
    tdm_load_2d_bf16((u32)(size_t)&As[buf][0][0],
                     A + (size_t)m0*K + kt*32,
                     32u, 128u, (u32)(K - kt*32), (u32)(M - m0), (u64)K,
                     1u, 3u, 3u);                      // pad 4 DWORDs per 16 -> stride 40
    // B tile: 32 contiguous k-elems x 128 n-rows of Bt, row stride K
    tdm_load_2d_bf16((u32)(size_t)&Bs[buf][0][0],
                     Bt + (size_t)n0*K + kt*32,
                     32u, 128u, (u32)(K - kt*32), (u32)(N - n0), (u64)K,
                     1u, 3u, 3u);                      // pad 4 DWORDs per 16 -> stride 40
  };
  if(wv==0) issue(0,0);
  for(int kt=0;kt<nk;kt++){
    const int buf=kt&1;
    if(wv==0){
      if(kt+1<nk){
        issue(kt+1, buf^1);
        __builtin_amdgcn_s_wait_tensorcnt(2);          // two oldest (tile kt) complete
      } else {
        __builtin_amdgcn_s_wait_tensorcnt(0);          // all complete
      }
    }
    __syncthreads();
    ABfrag af0, af1;
    loadA_rm(af0, &As[buf][wm*32   ][0], 40);
    loadA_rm(af1, &As[buf][wm*32+16][0], 40);
    #pragma unroll
    for(int ct=0;ct<4;ct++){
      ABfrag bfr;
      loadB_nk(bfr, &Bs[buf][wn*64+ct*16][0], 40);
      acc[0][ct]=wmma_bf(af0,bfr,acc[0][ct]);
      acc[1][ct]=wmma_bf(af1,bfr,acc[1][ct]);
    }
    __syncthreads();
  }
#else
  const int ar=tid>>1, ah=tid&1;    // both tiles: 128 rows x 2 halves of 16 elems
  for(int kt=0;kt<nk;kt++){
    const int buf=kt&1;
    uint4 a0=*(const uint4*)(A  + (size_t)(m0+ar)*K + kt*32 + ah*16);
    uint4 a1=*(const uint4*)(A  + (size_t)(m0+ar)*K + kt*32 + ah*16 + 8);
    uint4 b0=*(const uint4*)(Bt + (size_t)(n0+ar)*K + kt*32 + ah*16);
    uint4 b1=*(const uint4*)(Bt + (size_t)(n0+ar)*K + kt*32 + ah*16 + 8);
    if(kt+1<nk){
      __builtin_prefetch(A  + (size_t)(m0+ar)*K + (kt+1)*32 + ah*16, 0, 0);
      __builtin_prefetch(Bt + (size_t)(n0+ar)*K + (kt+1)*32 + ah*16, 0, 0);
    }
    __syncthreads();
    *(uint4*)&As[buf][ar][ah*16]   = a0;
    *(uint4*)&As[buf][ar][ah*16+8] = a1;
    *(uint4*)&Bs[buf][ar][ah*16]   = b0;
    *(uint4*)&Bs[buf][ar][ah*16+8] = b1;
    __syncthreads();
    ABfrag af0, af1;
    loadA_rm(af0, &As[buf][wm*32   ][0], 40);
    loadA_rm(af1, &As[buf][wm*32+16][0], 40);
    #pragma unroll
    for(int ct=0;ct<4;ct++){
      ABfrag bfr;
      loadB_nk(bfr, &Bs[buf][wn*64+ct*16][0], 40);
      acc[0][ct]=wmma_bf(af0,bfr,acc[0][ct]);
      acc[1][ct]=wmma_bf(af1,bfr,acc[1][ct]);
    }
  }
#endif

  const int hi=lane>>4, nn=lane&15;
  #pragma unroll
  for(int rt=0;rt<2;rt++){
    #pragma unroll
    for(int ct=0;ct<4;ct++){
      #pragma unroll
      for(int r=0;r<8;r++){
        const int row=m0+wm*32+rt*16+r+hi*8;
        const int col=n0+wn*64+ct*16+nn;
        float val=acc[rt][ct][r] + bias[col];
        if(epi==2)      val*=scale;
        else if(epi==3) val = 0.5f*val*(1.0f+erff(val*0.70710678118f));
        const size_t o=(size_t)row*N+col;
        if(epi==1) Cf[o]=val; else Cb[o]=f2bf(val);
      }
    }
  }
}

// ---------------------------------------------------------------------------
// Banded (Longformer) local attention, flash-style streaming softmax, WMMA for QK^T and PV.
__global__ __launch_bounds__(256) void attn_local_k(
    const u16* __restrict__ q, const u16* __restrict__ kk,
    const u16* __restrict__ vv, const float* __restrict__ amask,
    u16* __restrict__ attn)
{
  __shared__ u16 P[8][32][40];       // per-wave P bounce buffer (C-layout -> A-layout)
  const int bid=blockIdx.x;
  const int b=bid/96, hI=(bid/8)%12, n=bid%8;
  const int wv=threadIdx.x>>5, lane=threadIdx.x&31;
  const int hi=lane>>4, nn=lane&15;
  const int rowbase=n*256+wv*32;
  const size_t bstr=(size_t)b*2048;
  const v8f zero8 = {0.f,0.f,0.f,0.f,0.f,0.f,0.f,0.f};

  ABfrag aq[2][2];
  #pragma unroll
  for(int rt=0;rt<2;rt++)
    #pragma unroll
    for(int kh=0;kh<2;kh++)
      loadA_rm(aq[rt][kh], q + (bstr+rowbase+rt*16)*768 + hI*64 + kh*32, 768);

  v8f o[2][4];
  #pragma unroll
  for(int rt=0;rt<2;rt++)
    #pragma unroll
    for(int d=0;d<4;d++) o[rt][d]=zero8;
  float mrow[2][8], lrow[2][8];
  #pragma unroll
  for(int rt=0;rt<2;rt++)
    #pragma unroll
    for(int r=0;r<8;r++){ mrow[rt][r]=-1e30f; lrow[rt][r]=0.f; }

  for(int t=0;t<25;t++){             // 24 local 32-key tiles + 1 global-token tile
    const bool gt=(t==24);
    const int jbase=(n-1)*256+t*32;
    const int ctn = gt?1:2;
    v8f s[2][2];

    for(int ct=0;ct<ctn;ct++){
      ABfrag b0,b1;
      if(!gt){
        const int jp=jbase+ct*16+nn;
        const int jc=iclamp(jp,0,2047);
        const u16* kp = kk + (bstr+jc)*768 + hI*64 + hi*16;
        #pragma unroll
        for(int j=0;j<8;j++){ b0.w[j]=*(const u32*)(kp+2*j); b1.w[j]=*(const u32*)(kp+32+2*j); }
      } else {
        const u16* kp = kk + bstr*768 + hI*64 + hi*16;   // global key = token 0
        #pragma unroll
        for(int j=0;j<8;j++){
          const u32 x0=*(const u32*)(kp+2*j), x1=*(const u32*)(kp+32+2*j);
          b0.w[j]=(nn==0)?x0:0u; b1.w[j]=(nn==0)?x1:0u;
        }
      }
      #pragma unroll
      for(int rt=0;rt<2;rt++){
        v8f sv = wmma_bf(aq[rt][0], b0, zero8);
        sv     = wmma_bf(aq[rt][1], b1, sv);
        s[rt][ct]=sv;
      }
    }

    #pragma unroll
    for(int rt=0;rt<2;rt++){
      #pragma unroll
      for(int r=0;r<8;r++){
        const int qp=rowbase+rt*16+r+hi*8;
        float tmx=-1e30f;
        #pragma unroll
        for(int ct=0;ct<2;ct++){
          if(ct>=ctn) continue;
          float val=s[rt][ct][r];
          bool ok;
          if(gt){ ok = (nn==0) && (amask[bstr]>0.5f); }
          else {
            const int jp=jbase+ct*16+nn;
            ok = (jp>0)&&(jp<2048)&&(jp-qp<=256)&&(qp-jp<=256);
            if(ok) ok = amask[bstr+jp]>0.5f;
          }
          val = ok ? val : -1e9f;
          s[rt][ct][r]=val;
          tmx=fmaxf(tmx,val);
        }
        tmx=fmaxf(tmx,__shfl_xor(tmx,1,32));
        tmx=fmaxf(tmx,__shfl_xor(tmx,2,32));
        tmx=fmaxf(tmx,__shfl_xor(tmx,4,32));
        tmx=fmaxf(tmx,__shfl_xor(tmx,8,32));
        const float mo=mrow[rt][r];
        const float mn=fmaxf(mo,tmx);
        const float sc=__expf(mo-mn);
        mrow[rt][r]=mn;
        float rs=0.f;
        #pragma unroll
        for(int ct=0;ct<2;ct++){
          if(ct>=ctn) continue;
          const float pv=__expf(s[rt][ct][r]-mn);
          s[rt][ct][r]=pv; rs+=pv;
        }
        rs+=__shfl_xor(rs,1,32); rs+=__shfl_xor(rs,2,32);
        rs+=__shfl_xor(rs,4,32); rs+=__shfl_xor(rs,8,32);
        lrow[rt][r]=lrow[rt][r]*sc+rs;
        #pragma unroll
        for(int d=0;d<4;d++) o[rt][d][r]*=sc;
      }
    }

    #pragma unroll
    for(int rt=0;rt<2;rt++)
      #pragma unroll
      for(int ct=0;ct<2;ct++)
        #pragma unroll
        for(int r=0;r<8;r++)
          P[wv][rt*16+r+hi*8][ct*16+nn] = (ct<ctn)?f2bf(s[rt][ct][r]):(u16)0;

    ABfrag bvf[4];
    #pragma unroll
    for(int d=0;d<4;d++){
      #pragma unroll
      for(int j=0;j<8;j++){
        #pragma unroll
        for(int p2=0;p2<2;p2++){
          const int kr=2*j+hi*16+p2;
          const int jp = gt ? 0 : (jbase+kr);
          const int jc = iclamp(jp,0,2047);
          bvf[d].u[2*j+p2] = vv[(bstr+jc)*768 + hI*64 + d*16 + nn];
        }
      }
    }
    #pragma unroll
    for(int rt=0;rt<2;rt++){
      ABfrag ap;
      loadA_rm(ap, &P[wv][rt*16][0], 40);
      #pragma unroll
      for(int d=0;d<4;d++)
        o[rt][d]=wmma_bf(ap,bvf[d],o[rt][d]);
    }
  }

  #pragma unroll
  for(int rt=0;rt<2;rt++){
    #pragma unroll
    for(int r=0;r<8;r++){
      const float inv=1.0f/lrow[rt][r];
      const int qp=rowbase+rt*16+r+hi*8;
      #pragma unroll
      for(int d=0;d<4;d++)
        attn[(bstr+qp)*768 + hI*64 + d*16 + nn]=f2bf(o[rt][d][r]*inv);
    }
  }
}

// ---------------------------------------------------------------------------
__global__ __launch_bounds__(256) void attn_global_k(
    const float* __restrict__ h, const float* __restrict__ Wqg, const float* __restrict__ bqg,
    const u16* __restrict__ kg, const u16* __restrict__ vg,
    const float* __restrict__ amask, u16* __restrict__ attn)
{
  __shared__ float qg[64];
  __shared__ float pb[2048];
  __shared__ float red[256];
  const int b=blockIdx.x/12, hI=blockIdx.x%12;
  const int t=threadIdx.x;
  const size_t bstr=(size_t)b*2048;
  if(t<64){
    float acc=bqg[hI*64+t];
    const float* hr=h+bstr*768;
    for(int k2=0;k2<768;k2++) acc+=hr[k2]*Wqg[(size_t)k2*768+hI*64+t];
    qg[t]=acc*0.125f;
  }
  __syncthreads();
  float lmax=-1e30f;
  for(int s2=t;s2<2048;s2+=256){
    float acc=0.f;
    const u16* kr=kg+(bstr+s2)*768+hI*64;
    for(int d=0;d<64;d++) acc+=qg[d]*bf2f(kr[d]);
    if(amask[bstr+s2]<=0.5f) acc=-1e9f;
    pb[s2]=acc; lmax=fmaxf(lmax,acc);
  }
  red[t]=lmax; __syncthreads();
  for(int w2=128;w2>0;w2>>=1){ if(t<w2) red[t]=fmaxf(red[t],red[t+w2]); __syncthreads(); }
  const float mx=red[0]; __syncthreads();
  float ls=0.f;
  for(int s2=t;s2<2048;s2+=256){ const float e=__expf(pb[s2]-mx); pb[s2]=e; ls+=e; }
  red[t]=ls; __syncthreads();
  for(int w2=128;w2>0;w2>>=1){ if(t<w2) red[t]+=red[t+w2]; __syncthreads(); }
  const float inv=1.0f/red[0]; __syncthreads();
  const int d=t&63, seg=t>>6;
  float acc=0.f;
  for(int s2=seg*512;s2<seg*512+512;s2++) acc+=pb[s2]*bf2f(vg[(bstr+s2)*768+hI*64+d]);
  red[t]=acc; __syncthreads();
  if(t<64)
    attn[bstr*768 + hI*64 + t] = f2bf((red[t]+red[t+64]+red[t+128]+red[t+192])*inv);
}

// ---------------------------------------------------------------------------
__global__ __launch_bounds__(256) void resln_k(
    float* __restrict__ h, const float* __restrict__ x,
    const float* __restrict__ gam, const float* __restrict__ bet, u16* __restrict__ hb)
{
  __shared__ float r1[256], r2[256];
  const size_t base=(size_t)blockIdx.x*768;
  const int t=threadIdx.x;
  float v[3];
  #pragma unroll
  for(int i=0;i<3;i++){ const int idx=t+i*256; v[i]=h[base+idx]+x[base+idx]; }
  r1[t]=v[0]+v[1]+v[2];
  r2[t]=v[0]*v[0]+v[1]*v[1]+v[2]*v[2];
  __syncthreads();
  for(int w2=128;w2>0;w2>>=1){ if(t<w2){ r1[t]+=r1[t+w2]; r2[t]+=r2[t+w2]; } __syncthreads(); }
  const float mean=r1[0]*(1.0f/768.0f);
  const float var =r2[0]*(1.0f/768.0f)-mean*mean;
  const float rstd=rsqrtf(var+1e-5f);
  #pragma unroll
  for(int i=0;i<3;i++){
    const int idx=t+i*256;
    const float y=(v[i]-mean)*rstd*gam[idx]+bet[idx];
    h[base+idx]=y; hb[base+idx]=f2bf(y);
  }
}

__global__ __launch_bounds__(256) void embed_k(
    const int* __restrict__ ids, const float* __restrict__ ew, const float* __restrict__ ep,
    const float* __restrict__ gam, const float* __restrict__ bet,
    float* __restrict__ h, u16* __restrict__ hb)
{
  __shared__ float r1[256], r2[256];
  const int row=blockIdx.x; const int s=row&2047;
  const size_t base=(size_t)row*768;
  const int t=threadIdx.x;
  const int id=ids[row];
  float v[3];
  #pragma unroll
  for(int i=0;i<3;i++){
    const int idx=t+i*256;
    v[i]=ew[(size_t)id*768+idx]+ep[(size_t)s*768+idx];
  }
  r1[t]=v[0]+v[1]+v[2];
  r2[t]=v[0]*v[0]+v[1]*v[1]+v[2]*v[2];
  __syncthreads();
  for(int w2=128;w2>0;w2>>=1){ if(t<w2){ r1[t]+=r1[t+w2]; r2[t]+=r2[t+w2]; } __syncthreads(); }
  const float mean=r1[0]*(1.0f/768.0f);
  const float var =r2[0]*(1.0f/768.0f)-mean*mean;
  const float rstd=rsqrtf(var+1e-5f);
  #pragma unroll
  for(int i=0;i<3;i++){
    const int idx=t+i*256;
    const float y=(v[i]-mean)*rstd*gam[idx]+bet[idx];
    h[base+idx]=y; hb[base+idx]=f2bf(y);
  }
}

__global__ __launch_bounds__(256) void head_k(
    const float* __restrict__ h, const float* __restrict__ pW, const float* __restrict__ pb,
    const float* __restrict__ cW, const float* __restrict__ cb,
    const float* __restrict__ mW, const float* __restrict__ mb, float* __restrict__ out)
{
  __shared__ float pooled[4*768];
  __shared__ float lg[8];
  __shared__ float o16[16];
  const int t=threadIdx.x;
  for(int idx=t; idx<3072; idx+=256){
    const int bi=idx/768, d=idx%768;
    float acc=pb[d];
    const float* hr=h+(size_t)bi*2048*768;
    for(int k2=0;k2<768;k2++) acc+=hr[k2]*pW[(size_t)k2*768+d];
    pooled[idx]=tanhf(acc);
  }
  __syncthreads();
  if(t<8){
    const int bi=t>>1, cl=t&1;
    float acc=cb[cl];
    for(int k2=0;k2<768;k2++) acc+=pooled[bi*768+k2]*cW[k2*2+cl];
    lg[t]=acc;
  }
  __syncthreads();
  if(t<16){ float acc=mb[t]; for(int i=0;i<8;i++) acc+=lg[i]*mW[i*16+t]; o16[t]=acc; }
  __syncthreads();
  if(t==0){
    float mx=-1e30f; for(int i=0;i<16;i++) mx=fmaxf(mx,o16[i]);
    float sm=0.f;    for(int i=0;i<16;i++){ o16[i]=__expf(o16[i]-mx); sm+=o16[i]; }
    const float inv=1.0f/sm;
    for(int i=0;i<16;i++) out[i]=o16[i]*inv;
  }
}

// ---------------------------------------------------------------------------
extern "C" void kernel_launch(void* const* d_in, const int* in_sizes, int n_in,
                              void* d_out, int out_size, void* d_ws, size_t ws_size,
                              hipStream_t stream)
{
  (void)in_sizes; (void)n_in; (void)out_size; (void)ws_size;
  const int*   ids  =(const int*)  d_in[0];
  const float* amask=(const float*)d_in[1];
  const float* ew   =(const float*)d_in[2];
  const float* ep   =(const float*)d_in[3];
  const float* lne_s=(const float*)d_in[4];
  const float* lne_b=(const float*)d_in[5];
  const float* Wq =(const float*)d_in[6];
  const float* Wk =(const float*)d_in[7];
  const float* Wv =(const float*)d_in[8];
  const float* Wqg=(const float*)d_in[9];
  const float* Wkg=(const float*)d_in[10];
  const float* Wvg=(const float*)d_in[11];
  const float* Wo =(const float*)d_in[12];
  const float* bq =(const float*)d_in[13];
  const float* bk =(const float*)d_in[14];
  const float* bv =(const float*)d_in[15];
  const float* bqg=(const float*)d_in[16];
  const float* bkg=(const float*)d_in[17];
  const float* bvg=(const float*)d_in[18];
  const float* bo =(const float*)d_in[19];
  const float* ln1s=(const float*)d_in[20];
  const float* ln1b=(const float*)d_in[21];
  const float* Wi =(const float*)d_in[22];
  const float* bi =(const float*)d_in[23];
  const float* Wf =(const float*)d_in[24];
  const float* bfb=(const float*)d_in[25];
  const float* ln2s=(const float*)d_in[26];
  const float* ln2b=(const float*)d_in[27];
  const float* pW =(const float*)d_in[28];
  const float* pb =(const float*)d_in[29];
  const float* cW =(const float*)d_in[30];
  const float* cb =(const float*)d_in[31];
  const float* mW =(const float*)d_in[32];
  const float* mb =(const float*)d_in[33];
  float* out=(float*)d_out;

  char* wsp=(char*)d_ws; size_t off=0;
  auto alloc=[&](size_t bytes)->void*{ void* p=wsp+off; off+=(bytes+255)&~(size_t)255; return p; };
  const size_t MT=8192;                 // B*S tokens
  float* h    =(float*)alloc(MT*768*4);
  u16*   hb   =(u16*)  alloc(MT*768*2);
  u16*   qb   =(u16*)  alloc(MT*768*2);
  u16*   kb   =(u16*)  alloc(MT*768*2);
  u16*   vb   =(u16*)  alloc(MT*768*2);
  u16*   kgb  =(u16*)  alloc(MT*768*2);
  u16*   vgb  =(u16*)  alloc(MT*768*2);
  u16*   attnb=(u16*)  alloc(MT*768*2);
  u16*   ff1b =(u16*)  alloc(MT*3072*2);
  float* tmpf =(float*)alloc(MT*768*4);
  u16* wqb =(u16*)alloc(589824*2);      // all weight buffers hold W^T as [N][K] bf16
  u16* wkb =(u16*)alloc(589824*2);
  u16* wvb =(u16*)alloc(589824*2);
  u16* wkgb=(u16*)alloc(589824*2);
  u16* wvgb=(u16*)alloc(589824*2);
  u16* wob =(u16*)alloc(589824*2);
  u16* wib =(u16*)alloc(2359296*2);
  u16* wfb =(u16*)alloc(2359296*2);

  embed_k<<<8192,256,0,stream>>>(ids,ew,ep,lne_s,lne_b,h,hb);

  auto convtr=[&](const float* s,u16* d,int Kd,int Nd){
    f2bf_tr_k<<<(Nd/32)*(Kd/32),256,0,stream>>>(s,d,Kd,Nd);
  };
  auto gemm=[&](const u16*A,const u16*Bt,const float*bias,float*Cf,u16*Cb,int M,int N,int K,int epi,float sc){
    dim3 g(N/128,M/128);
    gemm_bf16_k<<<g,256,0,stream>>>(A,Bt,bias,Cf,Cb,M,N,K,epi,sc);
  };

  const size_t WDD=589824, WDF=2359296;
  for(int l=0;l<12;l++){
    convtr(Wq +l*WDD,wqb , 768, 768);
    convtr(Wk +l*WDD,wkb , 768, 768);
    convtr(Wv +l*WDD,wvb , 768, 768);
    convtr(Wkg+l*WDD,wkgb, 768, 768);
    convtr(Wvg+l*WDD,wvgb, 768, 768);
    convtr(Wo +l*WDD,wob , 768, 768);
    convtr(Wi +l*WDF,wib , 768,3072);
    convtr(Wf +l*WDF,wfb ,3072, 768);

    gemm(hb,wqb ,bq +l*768,nullptr,qb ,8192, 768, 768,2,0.125f);  // q = (h Wq + b)/sqrt(dh)
    gemm(hb,wkb ,bk +l*768,nullptr,kb ,8192, 768, 768,2,1.f);
    gemm(hb,wvb ,bv +l*768,nullptr,vb ,8192, 768, 768,2,1.f);
    gemm(hb,wkgb,bkg+l*768,nullptr,kgb,8192, 768, 768,2,1.f);
    gemm(hb,wvgb,bvg+l*768,nullptr,vgb,8192, 768, 768,2,1.f);

    attn_local_k <<<384,256,0,stream>>>(qb,kb,vb,amask,attnb);
    attn_global_k<<<48 ,256,0,stream>>>(h,Wqg+(size_t)l*WDD,bqg+l*768,kgb,vgb,amask,attnb);

    gemm(attnb,wob,bo+l*768,tmpf,nullptr,8192, 768, 768,1,1.f);
    resln_k<<<8192,256,0,stream>>>(h,tmpf,ln1s+l*768,ln1b+l*768,hb);

    gemm(hb  ,wib,bi +l*3072,nullptr,ff1b,8192,3072, 768,3,1.f); // GELU fused
    gemm(ff1b,wfb,bfb+l*768 ,tmpf,nullptr,8192, 768,3072,1,1.f);
    resln_k<<<8192,256,0,stream>>>(h,tmpf,ln2s+l*768,ln2b+l*768,hb);
  }

  head_k<<<1,256,0,stream>>>(h,pW,pb,cW,cb,mW,mb,out);
}
